// PackedStdScaler_14637248545461
// MI455X (gfx1250) — compile-verified, matrix operand order checked by reading, not run on hardware
//
#include <hip/hip_runtime.h>
#include <math.h>

typedef __attribute__((ext_vector_type(2)))  float    v2f;
typedef __attribute__((ext_vector_type(8)))  float    v8f;
typedef __attribute__((ext_vector_type(16))) _Float16 v16h;

#define S_LEN        2048   // sequence length (rows per batch)
#define D_LEN        128    // feature dim
#define GROUPS       64     // 8 sample_ids x 8 variate_ids
#define SLOT_STRIDE  4      // padded group slot: [sum_m, sum_tm, sum_t2m, pad]
#define MIN_SCALE    1e-5f

// ---------------------------------------------------------------- zero scratch
__global__ __launch_bounds__(256)
void PackedStdScaler_zero_ws(float* __restrict__ ws, int n) {
    for (int i = threadIdx.x; i < n; i += 256) ws[i] = 0.0f;
}

// ------------------------------------------------------- per-tile WMMA stats
// One wave (32 lanes) owns a 16-row x 128-col tile. Row sums of three
// matrices (m, t*m, t^2*m) are computed as A(16x4) x ones(4x16) WMMA chains.
__global__ __launch_bounds__(128)
void PackedStdScaler_stats(const float* __restrict__ target,
                           const unsigned char* __restrict__ mask,
                           const int* __restrict__ sid,
                           const int* __restrict__ vid,
                           float* __restrict__ ws,
                           int n_rows)
{
    __shared__ float s_at[4][16 * D_LEN];   // t * m
    __shared__ float s_am[4][16 * D_LEN];   // m (as float)

    const int lane = threadIdx.x & 31;
    const int wv   = threadIdx.x >> 5;
    const int tile = blockIdx.x * 4 + wv;
    if (tile * 16 >= n_rows) return;        // wave-uniform exit

    const long tile_base = (long)tile * 16 * D_LEN;

    // ---- coalesced load of the 16x128 tile into LDS (float4 / uchar4) ----
    #pragma unroll
    for (int it = 0; it < 16; ++it) {
        const int g   = it * 32 + lane;     // 512 float4 granules per tile
        const int row = g >> 5;             // 32 granules per row
        const int col = (g & 31) * 4;
        const float4 t4 = *(const float4*)(target + tile_base + row * D_LEN + col);
        const uchar4 m4 = *(const uchar4*)(mask   + tile_base + row * D_LEN + col);
        float4 fm, fa;
        fm.x = m4.x ? 1.0f : 0.0f;  fm.y = m4.y ? 1.0f : 0.0f;
        fm.z = m4.z ? 1.0f : 0.0f;  fm.w = m4.w ? 1.0f : 0.0f;
        fa.x = t4.x * fm.x;  fa.y = t4.y * fm.y;
        fa.z = t4.z * fm.z;  fa.w = t4.w * fm.w;
        *(float4*)&s_at[wv][row * D_LEN + col] = fa;
        *(float4*)&s_am[wv][row * D_LEN + col] = fm;
    }
    // Same-wave LDS store->load is in-order (DScnt); no barrier needed.

    v8f cm = {}; v8f ct = {}; v8f cs = {};
    const int row = lane & 15;
    const int kh  = lane >> 4;              // which K-half this lane supplies

#if __has_builtin(__builtin_amdgcn_wmma_f32_16x16x4_f32)
    // Full-precision f32 WMMA: A = 16x4 (v2f per lane), B = ones(4x16).
    const v2f b1 = {1.0f, 1.0f};
    #pragma unroll
    for (int kc = 0; kc < D_LEN / 4; ++kc) {
        const int col = kc * 4 + kh * 2;
        const float2 ft = *(const float2*)&s_at[wv][row * D_LEN + col];
        const float2 fm = *(const float2*)&s_am[wv][row * D_LEN + col];
        v2f aT = {ft.x, ft.y};
        v2f aM = {fm.x, fm.y};
        v2f aS = {ft.x * ft.x, ft.y * ft.y};   // (t*m)^2 == t^2*m for m in {0,1}
        cm = __builtin_amdgcn_wmma_f32_16x16x4_f32(false, aM, false, b1, (short)0, cm, false, false);
        ct = __builtin_amdgcn_wmma_f32_16x16x4_f32(false, aT, false, b1, (short)0, ct, false, false);
        cs = __builtin_amdgcn_wmma_f32_16x16x4_f32(false, aS, false, b1, (short)0, cs, false, false);
    }
#else
    // Fallback: codegen-confirmed f16 WMMA (A quantized to f16).
    v16h b1;
    #pragma unroll
    for (int j = 0; j < 16; ++j) b1[j] = (_Float16)1.0f;
    #pragma unroll
    for (int kc = 0; kc < D_LEN / 32; ++kc) {
        v16h aT, aM, aS;
        #pragma unroll
        for (int j = 0; j < 16; ++j) {
            const int col = kc * 32 + kh * 16 + j;
            const float t = s_at[wv][row * D_LEN + col];
            const float m = s_am[wv][row * D_LEN + col];
            aT[j] = (_Float16)t;
            aM[j] = (_Float16)m;
            aS[j] = (_Float16)(t * t);
        }
        cm = __builtin_amdgcn_wmma_f32_16x16x32_f16(false, aM, false, b1, (short)0, cm, false, false);
        ct = __builtin_amdgcn_wmma_f32_16x16x32_f16(false, aT, false, b1, (short)0, ct, false, false);
        cs = __builtin_amdgcn_wmma_f32_16x16x32_f16(false, aS, false, b1, (short)0, cs, false, false);
    }
#endif

    // D layout: lane 0 col0 holds rows 0..7 in VGPRs 0..7; lane 16 rows 8..15.
    if ((lane & 15) == 0) {
        const int rbase = (lane >> 4) * 8;
        #pragma unroll
        for (int r = 0; r < 8; ++r) {
            const int grow = tile * 16 + rbase + r;
            const int b    = grow / S_LEN;
            const int g    = sid[grow] * 8 + vid[grow];
            float* slot = ws + (size_t)(b * GROUPS + g) * SLOT_STRIDE;
            atomicAdd(slot + 0, cm[r]);
            atomicAdd(slot + 1, ct[r]);
            atomicAdd(slot + 2, cs[r]);
        }
    }
}

// ---------------------------------------------------------------- finalize
__global__ __launch_bounds__(256)
void PackedStdScaler_finalize(const int* __restrict__ sid,
                              const int* __restrict__ vid,
                              const float* __restrict__ ws,
                              float* __restrict__ out,
                              int n_rows)
{
    const int i = blockIdx.x * 256 + threadIdx.x;
    if (i >= n_rows) return;
    const int b = i / S_LEN;
    const int g = sid[i] * 8 + vid[i];
    const float* slot = ws + (size_t)(b * GROUPS + g) * SLOT_STRIDE;
    const float Sm = slot[0];      // tobs
    const float St = slot[1];      // sum t*m
    const float Ss = slot[2];      // sum t^2*m
    float loc = St / (Sm == 0.0f ? 1.0f : Sm);                 // safe_div
    const float den = Sm - 1.0f;                               // tobs - CORRECTION
    float var = (Ss - 2.0f * loc * St + loc * loc * Sm)
                / (den == 0.0f ? 1.0f : den);                  // safe_div
    float scale = sqrtf(var + MIN_SCALE);
    if (sid[i] == 0) { loc = 0.0f; scale = 1.0f; }             // padding
    out[i]          = loc;
    out[n_rows + i] = scale;
}

// ---------------------------------------------------------------- launch
extern "C" void kernel_launch(void* const* d_in, const int* in_sizes, int n_in,
                              void* d_out, int out_size, void* d_ws, size_t ws_size,
                              hipStream_t stream)
{
    const float*         target = (const float*)d_in[0];
    const unsigned char* mask   = (const unsigned char*)d_in[1]; // jnp.bool_: 1 byte
    const int*           sid    = (const int*)d_in[2];
    const int*           vid    = (const int*)d_in[3];
    float* ws  = (float*)d_ws;
    float* out = (float*)d_out;

    const int n_rows = in_sizes[2];            // B * S
    const int Bb     = n_rows / S_LEN;
    const int nslots = Bb * GROUPS * SLOT_STRIDE;

    PackedStdScaler_zero_ws<<<1, 256, 0, stream>>>(ws, nslots);

    const int n_tiles = n_rows / 16;           // 16 rows per wave
    PackedStdScaler_stats<<<(n_tiles + 3) / 4, 128, 0, stream>>>(
        target, mask, sid, vid, ws, n_rows);

    PackedStdScaler_finalize<<<(n_rows + 255) / 256, 256, 0, stream>>>(
        sid, vid, ws, out, n_rows);
}
